// TriangleMultiplicativeUpdate_2199023255766
// MI455X (gfx1250) — matrix operand order repaired and checked by reading, not compiled, not run on hardware
//
#include <hip/hip_runtime.h>
#include <stdint.h>

// ---------------------------------------------------------------------------
// Triangle multiplicative update (outgoing), B=1, N=512, Cz=Ch=128, MI455X.
//
// All GEMMs run on v_wmma_f32_16x16x32_bf16 (wave32). a/b/g/x intermediates
// are bf16 in workspace; a/b/g/x are stored channel-major [c][pos] so the
// per-channel triangle matmul reads contiguous rows. Entire a+b (128MB bf16)
// fits in the 192MB L2, so k-reuse in the triangle GEMM is L2-resident.
//
// Workspace layout (bytes):
//   [0,      163840)  W_all bf16 [640][128]  (ag,ap,bg,bp,g row-major over k)
//   [163840, 196608)  W_z   bf16 [128][128]
//   [196608, +64MB)   a bf16 [128][262144]
//   ( +64MB, +64MB)   b bf16 [128][262144]
//   ( +64MB, +64MB)   g bf16 [128][262144]
//   ( +64MB, +64MB)   x bf16 [128][262144]
// total ~268.6 MB
// ---------------------------------------------------------------------------

#define N_DIM 512
#define CZ    128
#define NPOS  (N_DIM * N_DIM)   // 262144

typedef __attribute__((ext_vector_type(16))) __bf16 v16bf;
typedef __attribute__((ext_vector_type(8)))  __bf16 v8bf;
typedef __attribute__((ext_vector_type(8)))  float  v8f;

__device__ __forceinline__ float sigmf(float x) {
  return 1.0f / (1.0f + __expf(-x));
}

__device__ __forceinline__ unsigned short f2bfu(float f) {
  unsigned u = __builtin_bit_cast(unsigned, f);
  u += 0x7FFFu + ((u >> 16) & 1u);   // round-to-nearest-even
  return (unsigned short)(u >> 16);
}
__device__ __forceinline__ __bf16 f2bf(float f) {
  unsigned short s = f2bfu(f);
  return __builtin_bit_cast(__bf16, s);
}
__device__ __forceinline__ float bf2f(__bf16 h) {
  unsigned u = ((unsigned)__builtin_bit_cast(unsigned short, h)) << 16;
  return __builtin_bit_cast(float, u);
}

// pack 8 floats -> 8 bf16 and store as one 16B chunk
__device__ __forceinline__ void store_bf8(__bf16* dst, const float v[8]) {
  uint4 u;
  u.x = (unsigned)f2bfu(v[0]) | ((unsigned)f2bfu(v[1]) << 16);
  u.y = (unsigned)f2bfu(v[2]) | ((unsigned)f2bfu(v[3]) << 16);
  u.z = (unsigned)f2bfu(v[4]) | ((unsigned)f2bfu(v[5]) << 16);
  u.w = (unsigned)f2bfu(v[6]) | ((unsigned)f2bfu(v[7]) << 16);
  *reinterpret_cast<uint4*>(dst) = u;
}

// Build a v16bf A/B fragment from two aligned 8-element chunks.
__device__ __forceinline__ v16bf pack16(v8bf lo, v8bf hi) {
  v16bf r;
#pragma unroll
  for (int i = 0; i < 8; ++i) { r[i] = lo[i]; r[8 + i] = hi[i]; }
  return r;
}

// ---------------------------------------------------------------------------
// K0: weights f32 -> bf16 workspace
// ---------------------------------------------------------------------------
__global__ __launch_bounds__(256) void k0_weights(
    const float* __restrict__ wag, const float* __restrict__ wap,
    const float* __restrict__ wbg, const float* __restrict__ wbp,
    const float* __restrict__ wg,  const float* __restrict__ wz,
    __bf16* __restrict__ wall, __bf16* __restrict__ wzb) {
  int idx = blockIdx.x * 256 + threadIdx.x;     // 0 .. 6*16384-1
  if (idx < 5 * 16384) {
    int g = idx >> 14, r = idx & 16383;
    const float* src = (g == 0) ? wag : (g == 1) ? wap : (g == 2) ? wbg
                      : (g == 3) ? wbp : wg;
    wall[idx] = f2bf(src[r]);
  } else {
    int j = idx - 5 * 16384;
    wzb[j] = f2bf(wz[j]);
  }
}

// ---------------------------------------------------------------------------
// K1: LayerNorm(z) -> 5 fused projections via WMMA -> a,b,g channel-major bf16
//   block = 256 threads (8 waves), 64 positions; output [64 x 640]
//   wave (mt, ng): M-tile mt (16 pos), covers 20 of the 40 N-tiles, paired so
//   sigmoid-gate partners (ag/ap, bg/bp) live in the same wave.
// ---------------------------------------------------------------------------
__global__ __launch_bounds__(256) void k1_proj(
    const float* __restrict__ z,    const float* __restrict__ mask,
    const float* __restrict__ lng,  const float* __restrict__ lnb,
    const float* __restrict__ bag,  const float* __restrict__ bap,
    const float* __restrict__ bbg,  const float* __restrict__ bbp,
    const float* __restrict__ bgg,
    const __bf16* __restrict__ wall,
    __bf16* __restrict__ aw, __bf16* __restrict__ bw, __bf16* __restrict__ gw) {
  __shared__ __align__(16) float  zf[64 * 132];    // padded f32 staging
  __shared__ __align__(16) __bf16 znb[64 * 136];   // 272B row stride
  __shared__ float mu[64];
  __shared__ float rs[64];

  const int tid = threadIdx.x;
  const int p0  = blockIdx.x * 64;

  // coalesced load of z tile [64][128] f32
  for (int t = tid; t < 64 * 32; t += 256) {
    int row = t >> 5, ch = t & 31;
    float4 v = *reinterpret_cast<const float4*>(z + (size_t)(p0 + row) * CZ + ch * 4);
    *reinterpret_cast<float4*>(&zf[row * 132 + ch * 4]) = v;
  }
  __syncthreads();

  if (tid < 64) {
    float s = 0.f, s2 = 0.f;
    for (int k = 0; k < CZ; ++k) { float v = zf[tid * 132 + k]; s += v; s2 += v * v; }
    float m = s * (1.0f / CZ);
    float var = s2 * (1.0f / CZ) - m * m;
    mu[tid] = m;
    rs[tid] = rsqrtf(var + 1e-5f);
  }
  __syncthreads();

  for (int t = tid; t < 64 * CZ; t += 256) {
    int row = t >> 7, k = t & 127;
    float v = (zf[row * 132 + k] - mu[row]) * rs[row] * lng[k] + lnb[k];
    znb[row * 136 + k] = f2bf(v);
  }
  __syncthreads();

  const int wv = tid >> 5, ln = tid & 31;
  const int mt = wv >> 1, ng = wv & 1;
  const int lhi = (ln >= 16) ? 1 : 0;
  const int l15 = ln & 15;

  const v8f zz = {0.f, 0.f, 0.f, 0.f, 0.f, 0.f, 0.f, 0.f};
  v8f acc[5][4];
#pragma unroll
  for (int g = 0; g < 5; ++g)
#pragma unroll
    for (int j = 0; j < 4; ++j) acc[g][j] = zz;

  const __bf16* arow = &znb[(mt * 16 + l15) * 136];
  const int aoff = lhi ? 8 : 0;     // A frag: lanes 16-31 hold K 8..15 / 24..31
  const int boff = lhi ? 8 : 0;     // B frag: lanes 16-31 hold K 8..15 / 24..31

  for (int kk = 0; kk < CZ; kk += 32) {
    v8bf alo = *reinterpret_cast<const v8bf*>(arow + kk + aoff);
    v8bf ahi = *reinterpret_cast<const v8bf*>(arow + kk + 16 + aoff);
    v16bf af = pack16(alo, ahi);
#pragma unroll
    for (int g = 0; g < 5; ++g) {
#pragma unroll
      for (int j = 0; j < 4; ++j) {
        int o = g * 128 + ng * 64 + j * 16 + l15;     // weight row (out channel)
        const __bf16* wrow = wall + (size_t)o * CZ + kk + boff;
        v8bf blo = *reinterpret_cast<const v8bf*>(wrow);
        v8bf bhi = *reinterpret_cast<const v8bf*>(wrow + 16);
        v16bf bf = pack16(blo, bhi);
        acc[g][j] = __builtin_amdgcn_wmma_f32_16x16x32_bf16(
            false, af, false, bf, (short)0, acc[g][j], false, false);
      }
    }
  }

  // epilogue: a = m*sig(ag)*ap ; b = m*sig(bg)*bp ; g = sig(glin)
  const int pb = p0 + mt * 16 + (lhi ? 8 : 0);   // 8 consecutive positions
  float mk[8];
  {
    float4 m0 = *reinterpret_cast<const float4*>(mask + pb);
    float4 m1 = *reinterpret_cast<const float4*>(mask + pb + 4);
    mk[0] = m0.x; mk[1] = m0.y; mk[2] = m0.z; mk[3] = m0.w;
    mk[4] = m1.x; mk[5] = m1.y; mk[6] = m1.z; mk[7] = m1.w;
  }
#pragma unroll
  for (int j = 0; j < 4; ++j) {
    int cw = ng * 64 + j * 16 + l15;    // channel 0..127
    float vag = bag[cw], vap = bap[cw], vbg = bbg[cw], vbp = bbp[cw], vgb = bgg[cw];
    float av[8], bv[8], gv[8];
#pragma unroll
    for (int v = 0; v < 8; ++v) {
      av[v] = mk[v] * sigmf(acc[0][j][v] + vag) * (acc[1][j][v] + vap);
      bv[v] = mk[v] * sigmf(acc[2][j][v] + vbg) * (acc[3][j][v] + vbp);
      gv[v] = sigmf(acc[4][j][v] + vgb);
    }
    store_bf8(aw + (size_t)cw * NPOS + pb, av);
    store_bf8(bw + (size_t)cw * NPOS + pb, bv);
    store_bf8(gw + (size_t)cw * NPOS + pb, gv);
  }
}

// ---------------------------------------------------------------------------
// K2: triangle einsum  x[c][i][j] = sum_k a[c][i][k] * b[c][j][k]
//   grid (8 i-tiles, 8 j-tiles, 128 channels); block = 8 waves, 64x64 output.
//   Fragments load directly from global (per-channel 1MB matrices sit in L2).
// ---------------------------------------------------------------------------
__global__ __launch_bounds__(256) void k2_tri(
    const __bf16* __restrict__ aw, const __bf16* __restrict__ bw,
    __bf16* __restrict__ xw) {
  __shared__ __align__(16) __bf16 xs[64 * 72];

  const int tid = threadIdx.x;
  const int wv = tid >> 5, ln = tid & 31;
  const int c  = blockIdx.z;
  const int i0 = blockIdx.x * 64, j0 = blockIdx.y * 64;
  const int mt = wv >> 1, ntb = (wv & 1) * 2;
  const int lhi = (ln >= 16) ? 1 : 0;
  const int l15 = ln & 15;

  const __bf16* Ac = aw + (size_t)c * NPOS;
  const __bf16* Bc = bw + (size_t)c * NPOS;
  const __bf16* arow  = Ac + (size_t)(i0 + mt * 16 + l15) * N_DIM;
  const __bf16* brow0 = Bc + (size_t)(j0 + ntb * 16 + l15) * N_DIM;
  const __bf16* brow1 = Bc + (size_t)(j0 + (ntb + 1) * 16 + l15) * N_DIM;
  const int aoff = lhi ? 8 : 0;
  const int boff = lhi ? 8 : 0;

  const v8f zz = {0.f, 0.f, 0.f, 0.f, 0.f, 0.f, 0.f, 0.f};
  v8f acc0 = zz, acc1 = zz;

  for (int k = 0; k < N_DIM; k += 32) {
    v16bf af = pack16(*reinterpret_cast<const v8bf*>(arow + k + aoff),
                      *reinterpret_cast<const v8bf*>(arow + k + 16 + aoff));
    v16bf b0 = pack16(*reinterpret_cast<const v8bf*>(brow0 + k + boff),
                      *reinterpret_cast<const v8bf*>(brow0 + k + boff + 16));
    v16bf b1 = pack16(*reinterpret_cast<const v8bf*>(brow1 + k + boff),
                      *reinterpret_cast<const v8bf*>(brow1 + k + boff + 16));
    acc0 = __builtin_amdgcn_wmma_f32_16x16x32_bf16(false, af, false, b0,
                                                   (short)0, acc0, false, false);
    acc1 = __builtin_amdgcn_wmma_f32_16x16x32_bf16(false, af, false, b1,
                                                   (short)0, acc1, false, false);
  }

  // stage tile in LDS, then coalesced bf16 stores to x[c][i][j]
#pragma unroll
  for (int v = 0; v < 8; ++v) {
    int il = mt * 16 + v + (lhi ? 8 : 0);
    xs[il * 72 + ntb * 16 + l15]       = f2bf(acc0[v]);
    xs[il * 72 + (ntb + 1) * 16 + l15] = f2bf(acc1[v]);
  }
  __syncthreads();
  for (int t = tid; t < 64 * 8; t += 256) {
    int il = t >> 3, ch = t & 7;
    uint4 v = *reinterpret_cast<const uint4*>(&xs[il * 72 + ch * 8]);
    *reinterpret_cast<uint4*>(xw + (size_t)c * NPOS +
                              (size_t)(i0 + il) * N_DIM + j0 + ch * 8) = v;
  }
}

// ---------------------------------------------------------------------------
// K3: out = (LN(x) @ wz^T + bz) * g   -- per 64-position tile
// ---------------------------------------------------------------------------
__global__ __launch_bounds__(256) void k3_out(
    const __bf16* __restrict__ xw, const __bf16* __restrict__ gw,
    const __bf16* __restrict__ wzb,
    const float* __restrict__ lng, const float* __restrict__ lnb,
    const float* __restrict__ bz, float* __restrict__ out) {
  __shared__ __align__(16) char smem[36352];
  __bf16* xf = reinterpret_cast<__bf16*>(smem);            // [128][72] bf16
  __bf16* xn = reinterpret_cast<__bf16*>(smem + 18432);    // [64][136] bf16
  float*  st = reinterpret_cast<float*>(smem + 35840);     // mu[64], rs[64]
  float*  ob = reinterpret_cast<float*>(smem);             // [64][128] f32 (reused)

  const int tid = threadIdx.x;
  const int p0  = blockIdx.x * 64;

  // load x[c][p0:p0+64] channel-major
  for (int t = tid; t < 128 * 8; t += 256) {
    int cc = t >> 3, ch = t & 7;
    uint4 v = *reinterpret_cast<const uint4*>(xw + (size_t)cc * NPOS + p0 + ch * 8);
    *reinterpret_cast<uint4*>(&xf[cc * 72 + ch * 8]) = v;
  }
  __syncthreads();

  if (tid < 64) {
    float s = 0.f, s2 = 0.f;
    for (int cc = 0; cc < CZ; ++cc) {
      float v = bf2f(xf[cc * 72 + tid]);
      s += v; s2 += v * v;
    }
    float m = s * (1.0f / CZ);
    float var = s2 * (1.0f / CZ) - m * m;
    st[tid] = m;
    st[64 + tid] = rsqrtf(var + 1e-5f);
  }
  __syncthreads();

  // normalized, transposed to [pos][c] bf16 for A fragments
  for (int t = tid; t < 64 * CZ; t += 256) {
    int p = t >> 7, cc = t & 127;
    float v = (bf2f(xf[cc * 72 + p]) - st[p]) * st[64 + p] * lng[cc] + lnb[cc];
    xn[p * 136 + cc] = f2bf(v);
  }
  __syncthreads();

  const int wv = tid >> 5, ln = tid & 31;
  const int mt = wv >> 1, ngq = wv & 1;
  const int lhi = (ln >= 16) ? 1 : 0;
  const int l15 = ln & 15;
  const int aoff = lhi ? 8 : 0;
  const int boff = lhi ? 8 : 0;

  const v8f zz = {0.f, 0.f, 0.f, 0.f, 0.f, 0.f, 0.f, 0.f};
  v8f acc[4];
#pragma unroll
  for (int j = 0; j < 4; ++j) acc[j] = zz;

  const __bf16* arow = &xn[(mt * 16 + l15) * 136];
  for (int kk = 0; kk < CZ; kk += 32) {
    v16bf af = pack16(*reinterpret_cast<const v8bf*>(arow + kk + aoff),
                      *reinterpret_cast<const v8bf*>(arow + kk + 16 + aoff));
#pragma unroll
    for (int j = 0; j < 4; ++j) {
      int o = ngq * 64 + j * 16 + l15;
      const __bf16* wrow = wzb + (size_t)o * CZ + kk + boff;
      v16bf bf = pack16(*reinterpret_cast<const v8bf*>(wrow),
                        *reinterpret_cast<const v8bf*>(wrow + 16));
      acc[j] = __builtin_amdgcn_wmma_f32_16x16x32_bf16(
          false, af, false, bf, (short)0, acc[j], false, false);
    }
  }
  __syncthreads();   // everyone done reading xf/xn before ob overwrites them

  const int pbl = mt * 16 + (lhi ? 8 : 0);   // local position base (8 consecutive)
#pragma unroll
  for (int j = 0; j < 4; ++j) {
    int co = ngq * 64 + j * 16 + l15;
    v8bf gvv = *reinterpret_cast<const v8bf*>(gw + (size_t)co * NPOS + p0 + pbl);
    float bzv = bz[co];
#pragma unroll
    for (int v = 0; v < 8; ++v)
      ob[(pbl + v) * 128 + co] = (acc[j][v] + bzv) * bf2f(gvv[v]);
  }
  __syncthreads();

  for (int t = tid; t < 64 * 32; t += 256) {
    int p = t >> 5, ch = t & 31;
    float4 v = *reinterpret_cast<const float4*>(&ob[p * 128 + ch * 4]);
    *reinterpret_cast<float4*>(out + (size_t)(p0 + p) * CZ + ch * 4) = v;
  }
}

// ---------------------------------------------------------------------------
extern "C" void kernel_launch(void* const* d_in, const int* in_sizes, int n_in,
                              void* d_out, int out_size, void* d_ws, size_t ws_size,
                              hipStream_t stream) {
  (void)in_sizes; (void)n_in; (void)out_size; (void)ws_size;

  const float* z    = (const float*)d_in[0];
  const float* mask = (const float*)d_in[1];
  const float* lnig = (const float*)d_in[2];
  const float* lnib = (const float*)d_in[3];
  const float* wag  = (const float*)d_in[4];
  const float* bag  = (const float*)d_in[5];
  const float* wap  = (const float*)d_in[6];
  const float* bap  = (const float*)d_in[7];
  const float* wbg  = (const float*)d_in[8];
  const float* bbg  = (const float*)d_in[9];
  const float* wbp  = (const float*)d_in[10];
  const float* bbp  = (const float*)d_in[11];
  const float* wg   = (const float*)d_in[12];
  const float* bg   = (const float*)d_in[13];
  const float* wz   = (const float*)d_in[14];
  const float* bz   = (const float*)d_in[15];
  const float* lnog = (const float*)d_in[16];
  const float* lnob = (const float*)d_in[17];
  float* out = (float*)d_out;

  char* ws = (char*)d_ws;
  const size_t OFF_WALL = 0;
  const size_t OFF_WZ   = 163840;
  const size_t OFF_A    = 196608;
  const size_t TENS     = (size_t)128 * NPOS * 2;   // 64MB each
  const size_t OFF_B    = OFF_A + TENS;
  const size_t OFF_G    = OFF_B + TENS;
  const size_t OFF_X    = OFF_G + TENS;

  __bf16* wall = (__bf16*)(ws + OFF_WALL);
  __bf16* wzb  = (__bf16*)(ws + OFF_WZ);
  __bf16* aw   = (__bf16*)(ws + OFF_A);
  __bf16* bw   = (__bf16*)(ws + OFF_B);
  __bf16* gw   = (__bf16*)(ws + OFF_G);
  __bf16* xw   = (__bf16*)(ws + OFF_X);

  k0_weights<<<384, 256, 0, stream>>>(wag, wap, wbg, wbp, wg, wz, wall, wzb);
  k1_proj<<<NPOS / 64, 256, 0, stream>>>(z, mask, lnig, lnib,
                                         bag, bap, bbg, bbp, bg,
                                         wall, aw, bw, gw);
  k2_tri<<<dim3(8, 8, 128), 256, 0, stream>>>(aw, bw, xw);
  k3_out<<<NPOS / 64, 256, 0, stream>>>(xw, gw, wzb, lnog, lnob, bz, out);
}
